// Attention_81037442941065
// MI455X (gfx1250) — matrix operand order repaired and checked by reading, not compiled, not run on hardware
//
#include <hip/hip_runtime.h>
#include <hip/hip_bf16.h>

// ---------------------------------------------------------------------------
// Dual attention (spatial + channel) for MI455X / gfx1250, bf16 WMMA pipeline.
// Round 4: GEMM templated on (N,K) so B-tile/k offsets fold into the 24-bit
// VGLOBAL immediate -> one A and one B base pointer, no scratch spills in the
// pipelined loop. Prefetches hoisted to the prologue.
// ---------------------------------------------------------------------------

typedef __bf16 bf16_t;
typedef __attribute__((ext_vector_type(16))) __bf16 v16bf;
typedef __attribute__((ext_vector_type(8)))  __bf16 v8bf;
typedef __attribute__((ext_vector_type(8)))  float  v8f;

#define DEV __device__ __forceinline__

static constexpr int B_   = 2;
static constexpr int N_   = 2048;
static constexpr int DIM_ = 1024;
static constexpr int H_   = 16;
static constexpr int D_   = 64;
static constexpr int INNER_ = H_ * D_;          // 1024
static constexpr int M_     = B_ * N_;          // 4096 rows

// ---------------------------------------------------------------------------
// WMMA helpers (layouts per CDNA5 ISA 7.12.2)
// ---------------------------------------------------------------------------
DEV v8f wmma_bf16(v16bf a, v16bf b, v8f c) {
  return __builtin_amdgcn_wmma_f32_16x16x32_bf16(
      /*neg_a=*/false, a, /*neg_b=*/false, b,
      /*c_mod=*/(short)0, c, /*reuse_a=*/false, /*reuse_b=*/false);
}

// Concatenate two v8bf chunks at p and p+gap into one WMMA operand register set.
DEV v16bf load_cat(const bf16_t* p, int gap) {
  v8bf lo = *(const v8bf*)(p);
  v8bf hi = *(const v8bf*)(p + gap);
  v16bf r;
#pragma unroll
  for (int e = 0; e < 8; ++e) { r[e] = lo[e]; r[e + 8] = hi[e]; }
  return r;
}

// A-operand 16x32: lane holds row M=lane&15; K chunks at kb=(lane>>4)*8 and kb+16.
DEV v16bf load_a16x32(const bf16_t* base, int ld, int row0, int k0, int lane) {
  const int m  = lane & 15;
  const int kb = (lane >> 4) * 8;
  return load_cat(base + (size_t)(row0 + m) * ld + k0 + kb, 16);
}

// B-operand 32x16 from K-contiguous storage BT[n][k]: lane holds col N=lane&15;
// K chunks at kb=(lane>>4)*8 and kb+16.
DEV v16bf load_b32x16(const bf16_t* baseT, int ld, int n0, int k0, int lane) {
  const int n  = lane & 15;
  const int kb = (lane >> 4) * 8;
  return load_cat(baseT + (size_t)(n0 + n) * ld + k0 + kb, 16);
}

// Reductions across the 16-lane N-group (wave32: masks 1,2,4,8 stay in-group).
DEV float xmax16(float v) {
  v = fmaxf(v, __shfl_xor(v, 1, 32));
  v = fmaxf(v, __shfl_xor(v, 2, 32));
  v = fmaxf(v, __shfl_xor(v, 4, 32));
  v = fmaxf(v, __shfl_xor(v, 8, 32));
  return v;
}
DEV float xsum16(float v) {
  v += __shfl_xor(v, 1, 32);
  v += __shfl_xor(v, 2, 32);
  v += __shfl_xor(v, 4, 32);
  v += __shfl_xor(v, 8, 32);
  return v;
}

// ---------------------------------------------------------------------------
// Stage 0: f32 -> bf16 conversions (plain and transposed)
// ---------------------------------------------------------------------------
__global__ void k_cvt_bf16(const float* __restrict__ s, bf16_t* __restrict__ d, size_t n) {
  for (size_t i = (size_t)blockIdx.x * blockDim.x + threadIdx.x; i < n;
       i += (size_t)gridDim.x * blockDim.x)
    d[i] = (bf16_t)s[i];
}

// dst[c*R + r] = src[r*C + c]   (weights stored K-major for B-operand loads)
__global__ void k_cvtT_bf16(const float* __restrict__ s, bf16_t* __restrict__ d,
                            int R, int C) {
  const size_t n = (size_t)R * C;
  for (size_t i = (size_t)blockIdx.x * blockDim.x + threadIdx.x; i < n;
       i += (size_t)gridDim.x * blockDim.x) {
    int r = (int)(i / C), c = (int)(i % C);
    d[(size_t)c * R + r] = (bf16_t)s[i];
  }
}

// ---------------------------------------------------------------------------
// Stage 1/4: C[M,NN] = A[M,KK] (bf16 row-major) x BT[NN,KK] (bf16 K-contiguous)
// Compile-time NN/KK: B-tile offsets t*16*KK and k offsets become instruction
// immediates, so the pipelined loop keeps exactly one A and one B base pointer.
// Wave = 16x64 strip, 2x-unrolled ping-pong k-loop (KK % 64 == 0).
// ---------------------------------------------------------------------------
template <int NN, int KK>
__global__ void k_gemm_wmma(const bf16_t* __restrict__ A, const bf16_t* __restrict__ BT,
                            bf16_t* __restrict__ Cb, float* __restrict__ Cf,
                            const float* __restrict__ bias, int M) {
  const int lane = threadIdx.x & 31;
  const int wid  = threadIdx.x >> 5;
  const int gw   = blockIdx.x * (blockDim.x >> 5) + wid;
  constexpr int ntg = NN >> 6;                  // N-tile groups of 64
  if (gw >= (M >> 4) * ntg) return;
  const int mt   = gw / ntg, ng = gw % ntg;
  const int row0 = mt * 16, col0 = ng * 64;

  // Per-lane base pointers; all tile/k displacements are compile-time.
  const bf16_t* ap = A  + (size_t)(row0 + (lane & 15)) * KK + (lane >> 4) * 8;
  const bf16_t* bp = BT + (size_t)(col0 + (lane & 15)) * KK + (lane >> 4) * 8;

  __builtin_prefetch(ap + 64, 0, 1);
  __builtin_prefetch(ap + 320, 0, 1);

  v8f c[4] = {};

  // prologue: ping buffer <- k=0
  v16bf a0 = load_cat(ap, 16);
  v16bf b0[4];
#pragma unroll
  for (int t = 0; t < 4; ++t) b0[t] = load_cat(bp + t * 16 * KK, 16);

  for (int k = 0; k < KK; k += 64) {
    // pong buffer <- k+32 (issued before consuming ping)
    v16bf a1 = load_cat(ap + k + 32, 16);
    v16bf b1[4];
#pragma unroll
    for (int t = 0; t < 4; ++t) b1[t] = load_cat(bp + t * 16 * KK + k + 32, 16);

#pragma unroll
    for (int t = 0; t < 4; ++t) c[t] = wmma_bf16(a0, b0[t], c[t]);

    // ping buffer <- k+64 (issued before consuming pong)
    if (k + 64 < KK) {
      a0 = load_cat(ap + k + 64, 16);
#pragma unroll
      for (int t = 0; t < 4; ++t) b0[t] = load_cat(bp + t * 16 * KK + k + 64, 16);
    }

#pragma unroll
    for (int t = 0; t < 4; ++t) c[t] = wmma_bf16(a1, b1[t], c[t]);
  }

  const int mh = (lane >> 4) * 8, nc = lane & 15;
  if (Cf) {
#pragma unroll
    for (int t = 0; t < 4; ++t)
#pragma unroll
      for (int r = 0; r < 8; ++r) {
        const size_t row = row0 + r + mh;
        const int    col = col0 + t * 16 + nc;
        Cf[row * NN + col] = c[t][r] + (bias ? bias[col] : 0.0f);
      }
  } else {
#pragma unroll
    for (int t = 0; t < 4; ++t)
#pragma unroll
      for (int r = 0; r < 8; ++r) {
        const size_t row = row0 + r + mh;
        const int    col = col0 + t * 16 + nc;
        Cb[row * NN + col] = (bf16_t)c[t][r];
      }
  }
}

// ---------------------------------------------------------------------------
// Stage 2: channel attention map  secmT[bh][j][i] = softmax_j(xh^T z2 /256)[i][j]
// One block (4 waves) per (b,h); wave w owns rows i in [16w,16w+16).
// ---------------------------------------------------------------------------
__global__ void k_channel_attn(const bf16_t* __restrict__ xh,
                               const bf16_t* __restrict__ z2,
                               bf16_t* __restrict__ secmT) {
  __shared__ __align__(16) bf16_t XT[D_ * 32];   // XT[d][n], n-chunk of 32
  __shared__ __align__(16) bf16_t ZT[D_ * 32];
  const int bh = blockIdx.x, b = bh >> 4, h = bh & 15;
  const int lane = threadIdx.x & 31, wid = threadIdx.x >> 5;
  const bf16_t* xp = xh + (size_t)b * N_ * INNER_ + h * D_;
  const bf16_t* zp = z2 + (size_t)b * N_ * INNER_ + h * D_;

  v8f c[4] = {};
  for (int nk = 0; nk < N_; nk += 32) {
    __syncthreads();
    for (int idx = threadIdx.x; idx < 32 * D_; idx += blockDim.x) {
      const int n = idx >> 6, d = idx & 63;
      XT[d * 32 + n] = xp[(size_t)(nk + n) * INNER_ + d];
      ZT[d * 32 + n] = zp[(size_t)(nk + n) * INNER_ + d];
    }
    __syncthreads();
    v16bf a = load_a16x32(XT, 32, wid * 16, 0, lane);   // rows = i, K = n
#pragma unroll
    for (int t = 0; t < 4; ++t) {
      v16bf bb = load_b32x16(ZT, 32, t * 16, 0, lane);  // cols = j, K = n
      c[t] = wmma_bf16(a, bb, c[t]);
    }
  }

  const float sc = 0.125f / 32.0f;                      // d^-0.5 / (n/d)
  const int mh = (lane >> 4) * 8, jc = lane & 15;
#pragma unroll
  for (int r = 0; r < 8; ++r) {
    float mx = -3.0e38f;
#pragma unroll
    for (int t = 0; t < 4; ++t) mx = fmaxf(mx, c[t][r] * sc);
    mx = xmax16(mx);
    float s = 0.0f;
    float p[4];
#pragma unroll
    for (int t = 0; t < 4; ++t) { p[t] = __expf(c[t][r] * sc - mx); s += p[t]; }
    s = xsum16(s);
    const float inv = 1.0f / s;
    const int i = wid * 16 + r + mh;
#pragma unroll
    for (int t = 0; t < 4; ++t) {
      const int j = t * 16 + jc;
      secmT[(size_t)bh * D_ * D_ + (size_t)j * D_ + i] = (bf16_t)(p[t] * inv);
    }
  }
}

// ---------------------------------------------------------------------------
// Stage 3: spatial flash-attention + fused channel output.
// grid = (N/64 row-blocks, B*H); block = 4 waves; wave = 16 query rows.
// ---------------------------------------------------------------------------
__global__ void k_spatial_attn(const bf16_t* __restrict__ z1,   // Q
                               const bf16_t* __restrict__ yh,   // K (and Y for out2)
                               const bf16_t* __restrict__ xh,   // V
                               const bf16_t* __restrict__ secmT,
                               bf16_t* __restrict__ attn) {
  __shared__ __align__(16) bf16_t VT[D_ * 32];     // VT[d][key], key-chunk of 32
  __shared__ __align__(16) bf16_t P[4][16 * 32];   // per-wave P transpose buffer
  const int bh = blockIdx.y, b = bh >> 4, h = bh & 15;
  const int lane = threadIdx.x & 31, wid = threadIdx.x >> 5;
  const size_t base = (size_t)b * N_ * INNER_ + h * D_;
  const bf16_t* Q  = z1 + base;
  const bf16_t* Kp = yh + base;
  const bf16_t* Vp = xh + base;
  const int row0 = blockIdx.x * 64 + wid * 16;     // query rows of this wave

  v16bf aq0 = load_a16x32(Q, INNER_, row0, 0, lane);
  v16bf aq1 = load_a16x32(Q, INNER_, row0, 32, lane);

  v8f o[4] = {};
  float m_i[8], l_i[8];
#pragma unroll
  for (int r = 0; r < 8; ++r) { m_i[r] = -3.0e38f; l_i[r] = 0.0f; }
  const float scale = 0.125f;                      // d^-0.5

  for (int kb = 0; kb < N_; kb += 32) {
    __syncthreads();
    for (int idx = threadIdx.x; idx < 32 * D_; idx += blockDim.x) {
      const int n = idx >> 6, d = idx & 63;
      VT[d * 32 + n] = Vp[(size_t)(kb + n) * INNER_ + d];
    }
    __syncthreads();

    // S tiles: keys kb..kb+15 (s0) and kb+16..kb+31 (s1), K-dim = d = 64
    v8f s0 = {}, s1 = {};
    s0 = wmma_bf16(aq0, load_b32x16(Kp, INNER_, kb,      0,  lane), s0);
    s0 = wmma_bf16(aq1, load_b32x16(Kp, INNER_, kb,      32, lane), s0);
    s1 = wmma_bf16(aq0, load_b32x16(Kp, INNER_, kb + 16, 0,  lane), s1);
    s1 = wmma_bf16(aq1, load_b32x16(Kp, INNER_, kb + 16, 32, lane), s1);

    // online softmax over the 32 keys of this chunk
#pragma unroll
    for (int r = 0; r < 8; ++r) {
      const float a0 = s0[r] * scale, a1 = s1[r] * scale;
      float mx = xmax16(fmaxf(a0, a1));
      const float mn   = fmaxf(m_i[r], mx);
      const float corr = __expf(m_i[r] - mn);
      const float p0 = __expf(a0 - mn), p1 = __expf(a1 - mn);
      l_i[r] = l_i[r] * corr + xsum16(p0 + p1);
      m_i[r] = mn;
#pragma unroll
      for (int t = 0; t < 4; ++t) o[t][r] *= corr;
      const int m = r + (lane >> 4) * 8;           // C-layout row
      P[wid][m * 32 + (lane & 15)]      = (bf16_t)p0;
      P[wid][m * 32 + 16 + (lane & 15)] = (bf16_t)p1;
    }
    asm volatile("s_wait_dscnt 0x0" ::: "memory"); // P stores -> P loads (same wave)

    v16bf ap = load_a16x32(&P[wid][0], 32, 0, 0, lane);
#pragma unroll
    for (int t = 0; t < 4; ++t) {
      v16bf bv = load_b32x16(VT, 32, t * 16, 0, lane);  // V[key][d] via VT[d][key]
      o[t] = wmma_bf16(ap, bv, o[t]);
    }
  }

  // finalize softmax normalization
#pragma unroll
  for (int r = 0; r < 8; ++r) {
    const float inv = 1.0f / l_i[r];
#pragma unroll
    for (int t = 0; t < 4; ++t) o[t][r] *= inv;
  }

  // fused channel output: O += Y_rows[16x64] @ secm[64x64]  (secm via secmT)
  v16bf ay0 = load_a16x32(Kp, INNER_, row0, 0, lane);
  v16bf ay1 = load_a16x32(Kp, INNER_, row0, 32, lane);
  const bf16_t* S = secmT + (size_t)bh * D_ * D_;
#pragma unroll
  for (int t = 0; t < 4; ++t) {
    o[t] = wmma_bf16(ay0, load_b32x16(S, D_, t * 16, 0,  lane), o[t]);
    o[t] = wmma_bf16(ay1, load_b32x16(S, D_, t * 16, 32, lane), o[t]);
  }

  const int mh = (lane >> 4) * 8, nc = lane & 15;
#pragma unroll
  for (int t = 0; t < 4; ++t)
#pragma unroll
    for (int r = 0; r < 8; ++r) {
      const size_t row = (size_t)b * N_ + row0 + r + mh;
      attn[row * INNER_ + h * D_ + t * 16 + nc] = (bf16_t)o[t][r];
    }
}

// ---------------------------------------------------------------------------
// Host: workspace carve-up + launch sequence (capture-safe, stream-ordered)
// ---------------------------------------------------------------------------
extern "C" void kernel_launch(void* const* d_in, const int* in_sizes, int n_in,
                              void* d_out, int out_size, void* d_ws, size_t ws_size,
                              hipStream_t stream) {
  (void)in_sizes; (void)n_in; (void)out_size; (void)ws_size;
  const float* x     = (const float*)d_in[0];
  const float* y     = (const float*)d_in[1];
  const float* z     = (const float*)d_in[2];
  const float* w_sa1 = (const float*)d_in[3];
  const float* w_sa2 = (const float*)d_in[4];
  const float* w_se1 = (const float*)d_in[5];
  const float* w_se2 = (const float*)d_in[6];
  const float* w_out = (const float*)d_in[7];
  const float* b_out = (const float*)d_in[8];
  float* out = (float*)d_out;

  char* ws = (char*)d_ws;
  size_t off = 0;
  auto take = [&](size_t bytes) -> void* {
    void* p = ws + off;
    off = (off + bytes + 255) & ~(size_t)255;
    return p;
  };
  const size_t actBytes = (size_t)M_ * DIM_ * sizeof(bf16_t);     // 8 MB
  const size_t wBytes   = (size_t)DIM_ * INNER_ * sizeof(bf16_t); // 2 MB
  bf16_t* xb    = (bf16_t*)take(actBytes);
  bf16_t* yb    = (bf16_t*)take(actBytes);
  bf16_t* zb    = (bf16_t*)take(actBytes);
  bf16_t* wsa1t = (bf16_t*)take(wBytes);
  bf16_t* wsa2t = (bf16_t*)take(wBytes);
  bf16_t* wse1t = (bf16_t*)take(wBytes);
  bf16_t* wse2t = (bf16_t*)take(wBytes);
  bf16_t* woutt = (bf16_t*)take(wBytes);
  bf16_t* z1p   = (bf16_t*)take(actBytes);
  bf16_t* yhp   = (bf16_t*)take(actBytes);
  bf16_t* xhp   = (bf16_t*)take(actBytes);
  bf16_t* z2p   = (bf16_t*)take(actBytes);
  bf16_t* attn  = (bf16_t*)take(actBytes);
  bf16_t* secmT = (bf16_t*)take((size_t)B_ * H_ * D_ * D_ * sizeof(bf16_t));

  // Stage 0: conversions
  const size_t nAct = (size_t)M_ * DIM_;
  k_cvt_bf16<<<2048, 256, 0, stream>>>(x, xb, nAct);
  k_cvt_bf16<<<2048, 256, 0, stream>>>(y, yb, nAct);
  k_cvt_bf16<<<2048, 256, 0, stream>>>(z, zb, nAct);
  k_cvtT_bf16<<<1024, 256, 0, stream>>>(w_sa1, wsa1t, DIM_, INNER_);
  k_cvtT_bf16<<<1024, 256, 0, stream>>>(w_sa2, wsa2t, DIM_, INNER_);
  k_cvtT_bf16<<<1024, 256, 0, stream>>>(w_se1, wse1t, DIM_, INNER_);
  k_cvtT_bf16<<<1024, 256, 0, stream>>>(w_se2, wse2t, DIM_, INNER_);
  k_cvtT_bf16<<<1024, 256, 0, stream>>>(w_out, woutt, INNER_, DIM_);

  // Stage 1: four projections (bf16 out)
  const int gemmBlocks = ((M_ / 16) * (INNER_ / 64) + 7) / 8;   // 512
  k_gemm_wmma<INNER_, DIM_><<<gemmBlocks, 256, 0, stream>>>(zb, wsa1t, z1p, nullptr, nullptr, M_);
  k_gemm_wmma<INNER_, DIM_><<<gemmBlocks, 256, 0, stream>>>(yb, wsa2t, yhp, nullptr, nullptr, M_);
  k_gemm_wmma<INNER_, DIM_><<<gemmBlocks, 256, 0, stream>>>(xb, wse1t, xhp, nullptr, nullptr, M_);
  k_gemm_wmma<INNER_, DIM_><<<gemmBlocks, 256, 0, stream>>>(zb, wse2t, z2p, nullptr, nullptr, M_);

  // Stage 2: channel attention maps
  k_channel_attn<<<B_ * H_, 128, 0, stream>>>(xhp, z2p, secmT);

  // Stage 3: spatial flash attention + fused channel output
  k_spatial_attn<<<dim3(N_ / 64, B_ * H_), 128, 0, stream>>>(z1p, yhp, xhp, secmT, attn);

  // Stage 4: output projection (f32 + bias) into d_out
  k_gemm_wmma<DIM_, INNER_><<<gemmBlocks, 256, 0, stream>>>(attn, woutt, nullptr, out, b_out, M_);
}
